// Quantize_78752520340269
// MI455X (gfx1250) — matrix operand order repaired and checked by reading, not compiled, not run on hardware
//
#include <hip/hip_runtime.h>
#include <hip/hip_bf16.h>

typedef __attribute__((ext_vector_type(16))) __bf16 v16bf;
typedef __attribute__((ext_vector_type(8)))  float  v8f;

#define B_   32
#define C_   256
#define H_   32
#define W_   32
#define K_   2048
#define N_   (B_*H_*W_)          // 32768
#define DCH  (C_/32)             // 8 d-chunks of 32
#define KT   (K_/16)             // 128 k-tiles of 16
#define ROWS 128                 // rows per block (8 waves x 16)
#define NBLK (N_/ROWS)           // 256
#define NPART (N_/32)            // 1024 diff partials

// ---- workspace layout (bytes) ----
#define EHI_OFF  0
#define ELO_OFF  (K_*C_*2)                  // 1 MB each
#define E2_OFF   (ELO_OFF + K_*C_*2)
#define IDX_OFF  (E2_OFF + K_*4)
#define HIST_OFF (IDX_OFF + N_*4)
#define PART_OFF (HIST_OFF + K_*4)

// ---- output layout (float elements), tuple order (q, diff, embed_ind, perplexity) ----
#define DIFF_OFF (B_*C_*H_*W_)              // 8388608
#define IND_OFF  (DIFF_OFF + 1)
#define PERP_OFF (IND_OFF + N_)             // 8421377

// Kernel 1: split embed into bf16 hi/lo, pre-swizzled into the WMMA B-matrix
// lane layout. ws element i = ((t*8 + c)*32 + l)*16 + e  maps to matrix
// element (d = c*32 + (e&7) + (l>=16 ? 8 : 0) + (e>=8 ? 16 : 0), k = t*16 + (l&15)).
__global__ void k_split_embed(const float* __restrict__ embed,
                              __bf16* __restrict__ ehi, __bf16* __restrict__ elo) {
    int i = blockIdx.x * 256 + threadIdx.x;   // 0 .. K*C-1
    int e = i & 15;
    int l = (i >> 4) & 31;
    int c = (i >> 9) & 7;
    int t = i >> 12;
    int d = c * 32 + (e & 7) + ((l >= 16) ? 8 : 0) + ((e >= 8) ? 16 : 0);
    int k = t * 16 + (l & 15);
    float v = embed[(size_t)d * K_ + k];
    __bf16 h = (__bf16)v;
    ehi[i] = h;
    elo[i] = (__bf16)(v - (float)h);
}

// Kernel 2: ||e_k||^2 per code + zero the histogram (re-zeroed every launch).
__global__ void k_e2(const float* __restrict__ embed,
                     float* __restrict__ e2, int* __restrict__ hist) {
    int k = blockIdx.x * 256 + threadIdx.x;
    float s = 0.f;
    for (int d = 0; d < C_; ++d) { float v = embed[(size_t)d * K_ + k]; s += v * v; }
    e2[k] = s;
    hist[k] = 0;
}

// Kernel 3: main WMMA distance + argmin. A lives in registers; B is async-staged.
__global__ void __launch_bounds__(256, 1)
k_vq_main(const float* __restrict__ x,
          const __bf16* __restrict__ ehi_g, const __bf16* __restrict__ elo_g,
          const float* __restrict__ e2, int* __restrict__ idxout,
          float* __restrict__ indf, int* __restrict__ hist) {
    extern __shared__ char smem[];
    // Phase 1 layout: A-hi [0,64K), A-lo [64K,128K)
    // Phase 2 layout (after A moved to registers): B double buffer aliases [0,32K)
    __bf16* Ahi = (__bf16*)smem;
    __bf16* Alo = Ahi + 8 * DCH * 32 * 16;

    const int tid  = threadIdx.x;
    const int lane = tid & 31;
    const int wv   = tid >> 5;
    const int nbase = blockIdx.x * ROWS;
    const unsigned lds_base = (unsigned)(size_t)smem;   // low 32 bits = LDS offset

    // ---- stage 128 rows of flat[n, d] = x[b, d, h, w] into A-swizzled LDS ----
    {
        int tr = tid & 127, tc = tid >> 7;          // row-local, d-phase
        int b = nbase >> 10, hw0 = nbase & 1023;    // 128-row blocks stay inside one image
        int m = tr & 15, w8 = tr >> 4;
        for (int dd = 0; dd < 128; ++dd) {
            int d = dd * 2 + tc;
            float v = x[((size_t)(b * C_ + d) << 10) + hw0 + tr];  // coalesced across lanes
            __bf16 h  = (__bf16)v;
            __bf16 lo = (__bf16)(v - (float)h);
            // ISA 16-bit A layout: lanes0-15 hold K {0..7,16..23}, lanes16-31 {8..15,24..31}
            int kk = d & 31, c = d >> 5;
            int l, e;
            if      (kk < 8)  { l = m;      e = kk;      }
            else if (kk < 16) { l = m + 16; e = kk - 8;  }
            else if (kk < 24) { l = m;      e = kk - 8;  }
            else              { l = m + 16; e = kk - 16; }
            int off = ((w8 * DCH + c) * 32 + l) * 16 + e;
            Ahi[off] = h;
            Alo[off] = lo;
        }
    }
    __syncthreads();

    // ---- pull this wave's A slice into registers (16 rows x 256 d, hi+lo) ----
    v16bf a_hi[DCH], a_lo[DCH];
    {
        const v16bf* Ah = (const v16bf*)Ahi;
        const v16bf* Al = (const v16bf*)Alo;
#pragma unroll
        for (int c = 0; c < DCH; ++c) {
            a_hi[c] = Ah[(wv * DCH + c) * 32 + lane];
            a_lo[c] = Al[(wv * DCH + c) * 32 + lane];
        }
    }
    __syncthreads();   // everyone done reading A -> B buffers may alias its LDS

    float bestv[8];
    int   bestk[8];
#pragma unroll
    for (int r = 0; r < 8; ++r) { bestv[r] = 3.4e38f; bestk[r] = 0; }

    // ---- async B stage: tile t = 8 KB hi + 8 KB lo; each thread moves 32 B of each.
    // GLOBAL_LOAD_ASYNC_TO_LDS_B128, GVS mode; IOFFSET is added to BOTH the global
    // and the LDS address, so offset:16 copies the second 16-byte half.
    auto stage_async = [&](int t, int buf) {
        unsigned go = (unsigned)t * 8192u + (unsigned)tid * 32u;           // byte off in ehi/elo
        unsigned lh = lds_base + (unsigned)buf * 16384u + (unsigned)tid * 32u;
        unsigned ll = lh + 8192u;
        asm volatile("global_load_async_to_lds_b128 %0, %1, %2"
                     :: "v"(lh), "v"(go), "s"(ehi_g) : "memory");
        asm volatile("global_load_async_to_lds_b128 %0, %1, %2 offset:16"
                     :: "v"(lh), "v"(go), "s"(ehi_g) : "memory");
        asm volatile("global_load_async_to_lds_b128 %0, %1, %2"
                     :: "v"(ll), "v"(go), "s"(elo_g) : "memory");
        asm volatile("global_load_async_to_lds_b128 %0, %1, %2 offset:16"
                     :: "v"(ll), "v"(go), "s"(elo_g) : "memory");
    };

    stage_async(0, 0);

    for (int t = 0; t < KT; ++t) {
        asm volatile("s_wait_asynccnt 0x0" ::: "memory");  // tile t resident in LDS
        __syncthreads();                                   // publish to all waves
        if (t + 1 < KT) stage_async(t + 1, (t + 1) & 1);

        const v16bf* Bh = (const v16bf*)(smem + (t & 1) * 16384);
        const v16bf* Bl = Bh + 256;                        // +8192 bytes

        v8f acc = {};                                      // f.e = hi.hi + hi.lo + lo.hi
#pragma unroll
        for (int c = 0; c < DCH; ++c) {
            v16bf bh = Bh[c * 32 + lane];
            v16bf bl = Bl[c * 32 + lane];
            acc = __builtin_amdgcn_wmma_f32_16x16x32_bf16(false, a_hi[c], false, bh, (short)0, acc, false, false);
            acc = __builtin_amdgcn_wmma_f32_16x16x32_bf16(false, a_hi[c], false, bl, (short)0, acc, false, false);
            acc = __builtin_amdgcn_wmma_f32_16x16x32_bf16(false, a_lo[c], false, bh, (short)0, acc, false, false);
        }

        int   kcol = t * 16 + (lane & 15);                 // C/D layout: N = lane%16
        float e2k  = e2[kcol];
#pragma unroll
        for (int r = 0; r < 8; ++r) {                      // row M = r + (lane>=16 ? 8 : 0)
            float dst = e2k - 2.0f * acc[r];               // ||f||^2 dropped (row-constant)
            if (dst < bestv[r]) { bestv[r] = dst; bestk[r] = kcol; }
        }
    }

    // reduce argmin across the 16 lanes sharing each row (stays in wave32 halves)
#pragma unroll
    for (int r = 0; r < 8; ++r) {
#pragma unroll
        for (int m2 = 1; m2 < 16; m2 <<= 1) {
            float ov = __shfl_xor(bestv[r], m2, 32);
            int   oi = __shfl_xor(bestk[r], m2, 32);
            if (ov < bestv[r] || (ov == bestv[r] && oi < bestk[r])) { bestv[r] = ov; bestk[r] = oi; }
        }
    }
    if ((lane & 15) == 0) {
        int rowoff = (lane >> 4) * 8;           // lane0 -> rows 0..7, lane16 -> rows 8..15
#pragma unroll
        for (int r = 0; r < 8; ++r) {
            int n = nbase + wv * 16 + rowoff + r;
            int kb = bestk[r];
            idxout[n] = kb;
            indf[n]   = (float)kb;
            atomicAdd(&hist[kb], 1);            // integer atomic: deterministic
        }
    }
}

// Kernel 4: exact fp32 gather q = embed[:, idx] + per-block partial MSE sums.
__global__ void k_gather_diff(const float* __restrict__ x, const float* __restrict__ embed,
                              const int* __restrict__ idx, float* __restrict__ q,
                              float* __restrict__ part) {
    __shared__ float red[256];
    int tid = threadIdx.x;
    int nb  = blockIdx.x * 32;                  // 32 consecutive n (same image)
    int b = nb >> 10, hw0 = nb & 1023;
    int j = tid & 31, g = tid >> 5;
    int n = nb + j;
    int ki = idx[n];
    float local = 0.f;
    for (int cc = 0; cc < 32; ++cc) {
        int c = cc * 8 + g;
        float ev = embed[(size_t)c * K_ + ki];  // 2 MB table: L2-resident gather
        size_t xo = ((size_t)(b * C_ + c) << 10) + hw0 + j;
        float xv = x[xo];
        q[xo] = ev;                             // coalesced across lanes
        float df = ev - xv;
        local += df * df;
    }
    red[tid] = local;
    __syncthreads();
    for (int s = 128; s > 0; s >>= 1) { if (tid < s) red[tid] += red[tid + s]; __syncthreads(); }
    if (tid == 0) part[blockIdx.x] = red[0];
}

// Kernel 5: finalize diff and perplexity.
__global__ void k_final(const float* __restrict__ part, const int* __restrict__ hist,
                        float* __restrict__ out) {
    __shared__ float red[256];
    int tid = threadIdx.x;
    float s = 0.f;
    for (int i = tid; i < NPART; i += 256) s += part[i];
    red[tid] = s;
    __syncthreads();
    for (int st = 128; st > 0; st >>= 1) { if (tid < st) red[tid] += red[tid + st]; __syncthreads(); }
    if (tid == 0) out[DIFF_OFF] = red[0] / (float)(B_*C_*H_*W_);
    __syncthreads();
    float sp = 0.f;
    for (int k = tid; k < K_; k += 256) {
        float p = (float)hist[k] * (1.0f / (float)N_);
        sp += p * logf(p + 1e-10f);
    }
    red[tid] = sp;
    __syncthreads();
    for (int st = 128; st > 0; st >>= 1) { if (tid < st) red[tid] += red[tid + st]; __syncthreads(); }
    if (tid == 0) out[PERP_OFF] = expf(-red[0]);
}

extern "C" void kernel_launch(void* const* d_in, const int* in_sizes, int n_in,
                              void* d_out, int out_size, void* d_ws, size_t ws_size,
                              hipStream_t stream) {
    const float* x     = (const float*)d_in[0];   // [32,256,32,32]
    const float* embed = (const float*)d_in[1];   // [256,2048]
    float* out = (float*)d_out;
    char*  ws  = (char*)d_ws;

    __bf16* ehi  = (__bf16*)(ws + EHI_OFF);
    __bf16* elo  = (__bf16*)(ws + ELO_OFF);
    float*  e2   = (float*)(ws + E2_OFF);
    int*    idx  = (int*)(ws + IDX_OFF);
    int*    hist = (int*)(ws + HIST_OFF);
    float*  part = (float*)(ws + PART_OFF);

    k_split_embed<<<(K_*C_)/256, 256, 0, stream>>>(embed, ehi, elo);
    k_e2<<<K_/256, 256, 0, stream>>>(embed, e2, hist);

    // dynamic LDS: A stage hi/lo (128 KB); B double-buffer (32 KB) aliases it
    size_t lds = (size_t)(8*DCH*32*16) * 2 * 2;
    k_vq_main<<<NBLK, 256, lds, stream>>>(x, ehi, elo, e2, idx, out + IND_OFF, hist);

    k_gather_diff<<<N_/32, 256, 0, stream>>>(x, embed, idx, out, part);
    k_final<<<1, 256, 0, stream>>>(part, hist, out);
}
